// Decoder_31473520345628
// MI455X (gfx1250) — matrix operand order repaired and checked by reading, not compiled, not run on hardware
//
#include <hip/hip_runtime.h>

// ---------------------------------------------------------------------------
// Persistent hierarchical-LSTM decoder for MI455X (gfx1250).
//   * bf16 WMMA (v_wmma_f32_16x16x32_bf16) for every matmul, weights packed
//     once per launch into a contiguous-K B-operand lane layout (k = 16*(lane/16) + i).
//   * All weights (~64 MB bf16) stay L2-resident (192 MB L2) -> the kernel is
//     recurrence-latency bound, so one persistent kernel + grid barriers.
//   * Activations staged to LDS per block (bank-conflict-free padded stride).
// ---------------------------------------------------------------------------

namespace {
constexpr int Bsz   = 32;
constexpr int LAT   = 512;
constexpr int CH    = 1024;
constexpr int CO    = 512;
constexpr int INP   = 389;
constexpr int Hd    = 1024;
constexpr int SEQn  = 1024;
constexpr int NSUB  = 8;
constexpr int STEPS = 128;
constexpr int G4    = 4096;              // 4*CH == 4*Hd
constexpr int Kc0   = LAT + CH;          // 1536 conductor L0 fused K
constexpr int Kc1   = 2 * CH;            // 2048 conductor L1 fused K
constexpr int Kd0   = 1952;              // (INP + CO) + Hd = 1925 -> pad 1952
constexpr int Kd1   = 2048;              // Hd + Hd
constexpr int KO    = 1024;              // output-proj K
constexpr int NOPAD = 400;               // INP padded to 16
constexpr int NBLK  = 32;
constexpr int NTHR  = 256;               // 8 waves/block, 256 waves total
constexpr int SECC  = 992;               // LDS K-section width (cols)
constexpr int SSTR  = SECC + 16;         // padded LDS row stride (bank-safe)
}

typedef __attribute__((ext_vector_type(16))) __bf16 v16bf;
typedef __attribute__((ext_vector_type(8)))  __bf16 v8bf;
typedef __attribute__((ext_vector_type(8)))  float  v8f;

struct DecP {
  const float *latent, *hdec, *cdec;
  const float *cb0, *cb1, *cOutb, *db0, *db1, *dOutb;
  const __bf16 *Wc0, *Wc1, *WcO, *Wd0, *Wd1, *WdO;
  __bf16 *Xc0, *Xc1, *Xd0, *Xd1;
  float *cc0, *cc1, *cd0, *cd1, *gates, *out;
  unsigned *bar;
};

// ---------------- grid-wide barrier (sense reversal) ----------------
__device__ __forceinline__ void gsync(unsigned* bar) {
  __threadfence();
  __syncthreads();
  if (threadIdx.x == 0) {
    unsigned s = __hip_atomic_load(&bar[1], __ATOMIC_RELAXED, __HIP_MEMORY_SCOPE_AGENT);
    if (__hip_atomic_fetch_add(&bar[0], 1u, __ATOMIC_ACQ_REL, __HIP_MEMORY_SCOPE_AGENT)
        == (unsigned)gridDim.x - 1u) {
      __hip_atomic_store(&bar[0], 0u, __ATOMIC_RELAXED, __HIP_MEMORY_SCOPE_AGENT);
      __hip_atomic_fetch_add(&bar[1], 1u, __ATOMIC_RELEASE, __HIP_MEMORY_SCOPE_AGENT);
    } else {
      while (__hip_atomic_load(&bar[1], __ATOMIC_ACQUIRE, __HIP_MEMORY_SCOPE_AGENT) == s)
        __builtin_amdgcn_s_sleep(2);
    }
  }
  __syncthreads();
  __threadfence();
}

__device__ __forceinline__ float sigf(float x) { return 1.0f / (1.0f + __expf(-x)); }

// ---------------- stage X (32 x sec bf16) into LDS ----------------
__device__ __forceinline__ void stage(__bf16* sX, const __bf16* Xg, int rowStride,
                                      int colBase, int sec) {
  const int cpr = sec >> 3;            // v8bf chunks per row
  const int nv  = 32 * cpr;
  for (int v = threadIdx.x; v < nv; v += blockDim.x) {
    int r  = v / cpr;
    int cv = v - r * cpr;
    *(v8bf*)(sX + r * SSTR + (cv << 3)) =
        *(const v8bf*)(Xg + (size_t)r * rowStride + colBase + (cv << 3));
  }
}

// A operand (contiguous K order, matches packed B): lane<16 -> K {0..15}, lane>=16 -> K {16..31}
__device__ __forceinline__ v16bf ld_a(const __bf16* row, int o0, int o1) {
  v8bf lo = *(const v8bf*)(row + o0);
  v8bf hi = *(const v8bf*)(row + o1);
  v16bf a;
#pragma unroll
  for (int i = 0; i < 8; ++i) { a[i] = lo[i]; a[i + 8] = hi[i]; }
  return a;
}

__device__ __forceinline__ void mma_section(const __bf16* sX, const __bf16* wSec,
                                            int nChunks, v8f& acc0, v8f& acc1, int lane) {
  const int mr = lane & 15;
  const int hiHalf = lane >> 4;
  const int o0 = hiHalf ? 16 : 0;
  const int o1 = hiHalf ? 24 : 8;
  const __bf16* r0 = sX + mr * SSTR;
  const __bf16* r1 = sX + (mr + 16) * SSTR;
  for (int kc = 0; kc < nChunks; ++kc) {
    __builtin_prefetch(wSec + (size_t)(kc + 2) * 512 + lane * 16, 0, 1);
    v16bf bm = *(const v16bf*)(wSec + (size_t)kc * 512 + lane * 16);
    v16bf a0 = ld_a(r0 + kc * 32, o0, o1);
    v16bf a1 = ld_a(r1 + kc * 32, o0, o1);
    acc0 = __builtin_amdgcn_wmma_f32_16x16x32_bf16(false, a0, false, bm, (short)0, acc0,
                                                   false, false);
    acc1 = __builtin_amdgcn_wmma_f32_16x16x32_bf16(false, a1, false, bm, (short)0, acc1,
                                                   false, false);
  }
}

// K-sectioned GEMM accumulate: OUT(32 x 16) tile pair for this wave.
__device__ __forceinline__ void gemm_acc(__bf16* sX, const __bf16* Xg, int rowStride,
                                         int kPad, const __bf16* Wp, int tile, bool active,
                                         int lane, v8f& a0, v8f& a1) {
  const int numKc = kPad >> 5;
  for (int base = 0; base < kPad; base += SECC) {
    int sec = kPad - base;
    if (sec > SECC) sec = SECC;
    __syncthreads();
    stage(sX, Xg, rowStride, base, sec);
    __syncthreads();
    if (active) {
      const __bf16* wSec = Wp + (((size_t)tile * numKc + (base >> 5)) << 9);
      mma_section(sX, wSec, sec >> 5, a0, a1, lane);
    }
  }
}

__device__ __forceinline__ void gates_gemm(__bf16* sX, const __bf16* Xg, int rowStride,
                                           int kPad, const __bf16* Wp, const float* bias,
                                           float* gates, int wave, int lane) {
  const int col = wave * 16 + (lane & 15);
  const float bv = bias[col];
  v8f a0 = {bv, bv, bv, bv, bv, bv, bv, bv};
  v8f a1 = a0;
  gemm_acc(sX, Xg, rowStride, kPad, Wp, wave, true, lane, a0, a1);
  const int r0 = (lane < 16) ? 0 : 8;
#pragma unroll
  for (int r = 0; r < 8; ++r) {
    gates[(size_t)(r0 + r) * G4 + col]      = a0[r];
    gates[(size_t)(r0 + r + 16) * G4 + col] = a1[r];
  }
}

// LSTM elementwise: i,f,g,o gate order; writes h (bf16) into one or two X slots.
__device__ __forceinline__ void cell(const float* gates, float* cSt, __bf16* h1d, int s1,
                                     __bf16* h2d, int s2, int gtid, int nthr) {
  for (int i = gtid; i < Bsz * 1024; i += nthr) {
    int b = i >> 10, j = i & 1023;
    const float* g = gates + (size_t)b * G4;
    float gi = g[j], gf = g[1024 + j], gg = g[2048 + j], go = g[3072 + j];
    float c = sigf(gf) * cSt[i] + sigf(gi) * tanhf(gg);
    float h = sigf(go) * tanhf(c);
    cSt[i] = c;
    __bf16 hb = (__bf16)h;
    h1d[(size_t)b * s1 + j] = hb;
    if (h2d) h2d[(size_t)b * s2 + j] = hb;
  }
}

// ---------------- weight packer: fp32 [W1|W2] -> contiguous-K WMMA-B bf16 layout ----
__global__ void pack_w(const float* __restrict__ W1, int K1, const float* __restrict__ W2,
                       int K2, int Nsrc, int Npad, int Kpad, __bf16* __restrict__ dst) {
  size_t idx = (size_t)blockIdx.x * blockDim.x + threadIdx.x;
  size_t total = (size_t)Npad * Kpad;
  if (idx >= total) return;
  int n = (int)(idx / Kpad);
  int k = (int)(idx - (size_t)n * Kpad);
  float v = 0.f;
  if (n < Nsrc) {
    if (k < K1)               v = W1[(size_t)n * K1 + k];
    else if (W2 && k < K1+K2) v = W2[(size_t)n * K2 + (k - K1)];
  }
  int t = n >> 4, nl = n & 15, kc = k >> 5, kk = k & 31;
  int lane2 = ((kk >= 16) ? 16 : 0) + nl;
  int i2 = kk & 15;
  size_t off = ((((size_t)t * (Kpad >> 5) + kc) << 5) + lane2) * 16 + i2;
  dst[off] = (__bf16)v;
}

__global__ void init_bar(unsigned* bar) {
  if (threadIdx.x == 0) { bar[0] = 0u; bar[1] = 0u; }
}

// ---------------- persistent decoder ----------------
__global__ __launch_bounds__(NTHR) void decoder_persistent(DecP p) {
  __shared__ __bf16 sX[32 * SSTR];
  const int lane = threadIdx.x & 31;
  const int gtid = blockIdx.x * blockDim.x + threadIdx.x;
  const int nthr = gridDim.x * blockDim.x;
  const int wave = gtid >> 5;

  // ---- init: zero states / pads, load latent into conductor X ----
  for (int i = gtid; i < Bsz * Kd0; i += nthr) p.Xd0[i] = (__bf16)0.f;
  for (int i = gtid; i < Bsz * Kd1; i += nthr) p.Xd1[i] = (__bf16)0.f;
  for (int i = gtid; i < Bsz * Kc1; i += nthr) p.Xc1[i] = (__bf16)0.f;
  for (int i = gtid; i < Bsz * CH; i += nthr) {      // hc0 slot of Xc0
    int b = i >> 10, c = i & 1023;
    p.Xc0[(size_t)b * Kc0 + LAT + c] = (__bf16)0.f;
  }
  for (int i = gtid; i < Bsz * LAT; i += nthr) {     // latent slot of Xc0
    int b = i >> 9, c = i & 511;
    p.Xc0[(size_t)b * Kc0 + c] = (__bf16)p.latent[i];
  }
  for (int i = gtid; i < Bsz * CH; i += nthr) { p.cc0[i] = 0.f; p.cc1[i] = 0.f; }
  gsync(p.bar);

  for (int sub = 0; sub < NSUB; ++sub) {
    // fresh per-subsequence decoder state
    const float* h0i = p.hdec + ((size_t)sub * 2 + 0) * Bsz * Hd;
    const float* h1i = p.hdec + ((size_t)sub * 2 + 1) * Bsz * Hd;
    const float* c0i = p.cdec + ((size_t)sub * 2 + 0) * Bsz * Hd;
    const float* c1i = p.cdec + ((size_t)sub * 2 + 1) * Bsz * Hd;
    for (int i = gtid; i < Bsz * Hd; i += nthr) {
      int b = i >> 10, j = i & 1023;
      p.Xd0[(size_t)b * Kd0 + (INP + CO) + j] = (__bf16)h0i[i];
      p.Xd1[(size_t)b * Kd1 + Hd + j]         = (__bf16)h1i[i];
      p.cd0[i] = c0i[i];
      p.cd1[i] = c1i[i];
    }
    gsync(p.bar);

    // ---- conductor step (state carried across subsequences) ----
    gates_gemm(sX, p.Xc0, Kc0, Kc0, p.Wc0, p.cb0, p.gates, wave, lane);
    gsync(p.bar);
    cell(p.gates, p.cc0, p.Xc0 + LAT, Kc0, p.Xc1, Kc1, gtid, nthr);
    gsync(p.bar);
    gates_gemm(sX, p.Xc1, Kc1, Kc1, p.Wc1, p.cb1, p.gates, wave, lane);
    gsync(p.bar);
    cell(p.gates, p.cc1, p.Xc1 + CH, Kc1, nullptr, 0, gtid, nthr);
    gsync(p.bar);
    {  // emb = tanh(hc1 @ cOut_W.T + b) -> bf16 into Xd0[:, INP:INP+CO]
      bool act = wave < (CO / 16);
      int col = wave * 16 + (lane & 15);
      float bv = act ? p.cOutb[col] : 0.f;
      v8f a0 = {bv, bv, bv, bv, bv, bv, bv, bv};
      v8f a1 = a0;
      gemm_acc(sX, p.Xc1 + CH, Kc1, KO, p.WcO, wave, act, lane, a0, a1);
      if (act) {
        int r0 = (lane < 16) ? 0 : 8;
#pragma unroll
        for (int r = 0; r < 8; ++r) {
          p.Xd0[(size_t)(r0 + r) * Kd0 + INP + col]      = (__bf16)tanhf(a0[r]);
          p.Xd0[(size_t)(r0 + r + 16) * Kd0 + INP + col] = (__bf16)tanhf(a1[r]);
        }
      }
    }
    gsync(p.bar);

    // ---- 128 note steps ----
    for (int step = 0; step < STEPS; ++step) {
      gates_gemm(sX, p.Xd0, Kd0, Kd0, p.Wd0, p.db0, p.gates, wave, lane);
      gsync(p.bar);
      cell(p.gates, p.cd0, p.Xd0 + (INP + CO), Kd0, p.Xd1, Kd1, gtid, nthr);
      gsync(p.bar);
      gates_gemm(sX, p.Xd1, Kd1, Kd1, p.Wd1, p.db1, p.gates, wave, lane);
      gsync(p.bar);
      cell(p.gates, p.cd1, p.Xd1 + Hd, Kd1, nullptr, 0, gtid, nthr);
      gsync(p.bar);
      {  // prev = tanh(h1 @ dOut_W.T + b); emit output + feedback
        bool act = wave < (NOPAD / 16);
        int col = wave * 16 + (lane & 15);
        float bv = (act && col < INP) ? p.dOutb[col] : 0.f;
        v8f a0 = {bv, bv, bv, bv, bv, bv, bv, bv};
        v8f a1 = a0;
        gemm_acc(sX, p.Xd1 + Hd, Kd1, KO, p.WdO, wave, act, lane, a0, a1);
        if (act && col < INP) {
          int t = sub * STEPS + step;
          int r0 = (lane < 16) ? 0 : 8;
#pragma unroll
          for (int r = 0; r < 8; ++r) {
            float v0 = tanhf(a0[r]);
            float v1 = tanhf(a1[r]);
            p.out[((size_t)(r0 + r) * SEQn + t) * INP + col]      = v0;
            p.out[((size_t)(r0 + r + 16) * SEQn + t) * INP + col] = v1;
            p.Xd0[(size_t)(r0 + r) * Kd0 + col]      = (__bf16)v0;
            p.Xd0[(size_t)(r0 + r + 16) * Kd0 + col] = (__bf16)v1;
          }
        }
      }
      gsync(p.bar);
    }
  }
}

// ---------------------------------------------------------------------------
extern "C" void kernel_launch(void* const* d_in, const int* in_sizes, int n_in,
                              void* d_out, int out_size, void* d_ws, size_t ws_size,
                              hipStream_t stream) {
  (void)in_sizes; (void)n_in; (void)out_size; (void)ws_size;
  const float* latent = (const float*)d_in[0];
  // d_in[1] = target (unused; teacher forcing off)
  const float* hdec  = (const float*)d_in[2];
  const float* cdec  = (const float*)d_in[3];
  const float* cW_ih0 = (const float*)d_in[4];
  const float* cW_hh0 = (const float*)d_in[5];
  const float* cb0    = (const float*)d_in[6];
  const float* cW_ih1 = (const float*)d_in[7];
  const float* cW_hh1 = (const float*)d_in[8];
  const float* cb1    = (const float*)d_in[9];
  const float* cOut_W = (const float*)d_in[10];
  const float* cOut_b = (const float*)d_in[11];
  const float* dW_ih0 = (const float*)d_in[12];
  const float* dW_hh0 = (const float*)d_in[13];
  const float* db0    = (const float*)d_in[14];
  const float* dW_ih1 = (const float*)d_in[15];
  const float* dW_hh1 = (const float*)d_in[16];
  const float* db1    = (const float*)d_in[17];
  const float* dOut_W = (const float*)d_in[18];
  const float* dOut_b = (const float*)d_in[19];

  // ---- carve workspace (~66 MB; 256B aligned regions) ----
  char* ws = (char*)d_ws;
  size_t off = 0;
  auto carve = [&](size_t bytes) -> void* {
    off = (off + 255) & ~(size_t)255;
    void* r = ws + off;
    off += bytes;
    return r;
  };
  __bf16* Wc0p = (__bf16*)carve((size_t)G4 * Kc0 * 2);
  __bf16* Wc1p = (__bf16*)carve((size_t)G4 * Kc1 * 2);
  __bf16* WcOp = (__bf16*)carve((size_t)CO * KO * 2);
  __bf16* Wd0p = (__bf16*)carve((size_t)G4 * Kd0 * 2);
  __bf16* Wd1p = (__bf16*)carve((size_t)G4 * Kd1 * 2);
  __bf16* WdOp = (__bf16*)carve((size_t)NOPAD * KO * 2);
  float*  gates = (float*)carve((size_t)Bsz * G4 * 4);
  __bf16* Xc0  = (__bf16*)carve((size_t)Bsz * Kc0 * 2);
  __bf16* Xc1  = (__bf16*)carve((size_t)Bsz * Kc1 * 2);
  __bf16* Xd0  = (__bf16*)carve((size_t)Bsz * Kd0 * 2);
  __bf16* Xd1  = (__bf16*)carve((size_t)Bsz * Kd1 * 2);
  float* cc0 = (float*)carve((size_t)Bsz * CH * 4);
  float* cc1 = (float*)carve((size_t)Bsz * CH * 4);
  float* cd0 = (float*)carve((size_t)Bsz * Hd * 4);
  float* cd1 = (float*)carve((size_t)Bsz * Hd * 4);
  unsigned* bar = (unsigned*)carve(2 * sizeof(unsigned));

  auto packGrid = [](size_t total) { return dim3((unsigned)((total + 255) / 256)); };
  pack_w<<<packGrid((size_t)G4 * Kc0), 256, 0, stream>>>(cW_ih0, LAT, cW_hh0, CH, G4, G4, Kc0, Wc0p);
  pack_w<<<packGrid((size_t)G4 * Kc1), 256, 0, stream>>>(cW_ih1, CH, cW_hh1, CH, G4, G4, Kc1, Wc1p);
  pack_w<<<packGrid((size_t)CO * KO), 256, 0, stream>>>(cOut_W, CH, nullptr, 0, CO, CO, KO, WcOp);
  pack_w<<<packGrid((size_t)G4 * Kd0), 256, 0, stream>>>(dW_ih0, INP + CO, dW_hh0, Hd, G4, G4, Kd0, Wd0p);
  pack_w<<<packGrid((size_t)G4 * Kd1), 256, 0, stream>>>(dW_ih1, Hd, dW_hh1, Hd, G4, G4, Kd1, Wd1p);
  pack_w<<<packGrid((size_t)NOPAD * KO), 256, 0, stream>>>(dOut_W, Hd, nullptr, 0, INP, NOPAD, KO, WdOp);
  init_bar<<<1, 32, 0, stream>>>(bar);

  DecP p;
  p.latent = latent; p.hdec = hdec; p.cdec = cdec;
  p.cb0 = cb0; p.cb1 = cb1; p.cOutb = cOut_b;
  p.db0 = db0; p.db1 = db1; p.dOutb = dOut_b;
  p.Wc0 = Wc0p; p.Wc1 = Wc1p; p.WcO = WcOp;
  p.Wd0 = Wd0p; p.Wd1 = Wd1p; p.WdO = WdOp;
  p.Xc0 = Xc0; p.Xc1 = Xc1; p.Xd0 = Xd0; p.Xd1 = Xd1;
  p.cc0 = cc0; p.cc1 = cc1; p.cd0 = cd0; p.cd1 = cd1;
  p.gates = gates; p.out = (float*)d_out; p.bar = bar;

  decoder_persistent<<<NBLK, NTHR, 0, stream>>>(p);
}
